// SCSA_11218454577406
// MI455X (gfx1250) — matrix operand order repaired and checked by reading, not compiled, not run on hardware
//
#include <hip/hip_runtime.h>
#include <hip/hip_bf16.h>
#include <math.h>

#define BATCH   32
#define DIM     512
#define HEIGHT  56
#define WIDTH   56
#define PLANE   (HEIGHT*WIDTH)   // 3136
#define GC      128
#define HEADS   8
#define HEAD_DIM 64
#define LL      64               // 8*8 pooled spatial
#define WIN     7
#define EPS_F   1e-5f

typedef __attribute__((ext_vector_type(16))) _Float16 v16h;
typedef __attribute__((ext_vector_type(8)))  float    v8f;
typedef __attribute__((ext_vector_type(4)))  float    f4;   // vector type for NT builtins

// -------- Kernel 1: row/col means of each (b,c) plane (b128 staged in LDS) --------
__global__ void k1_axis_means(const float* __restrict__ x,
                              float* __restrict__ xh, float* __restrict__ xw) {
  __shared__ __attribute__((aligned(16))) float sm[PLANE];
  const int plane = blockIdx.x;                    // b*DIM + c
  const f4* src4 = reinterpret_cast<const f4*>(x + (size_t)plane * PLANE);
  f4* sm4 = reinterpret_cast<f4*>(sm);
  for (int i = threadIdx.x; i < PLANE/4; i += blockDim.x)
    sm4[i] = src4[i];                              // global b128 -> ds b128
  __syncthreads();
  const int t = threadIdx.x;
  if (t < HEIGHT) {
    float s = 0.f;
    #pragma unroll 8
    for (int w = 0; w < WIDTH; ++w) s += sm[t*WIDTH + w];
    xh[(size_t)plane*HEIGHT + t] = s * (1.f/WIDTH);
  } else if (t >= 64 && t < 64 + WIDTH) {
    const int w = t - 64;
    float s = 0.f;
    #pragma unroll 8
    for (int h = 0; h < HEIGHT; ++h) s += sm[h*WIDTH + w];
    xw[(size_t)plane*WIDTH + w] = s * (1.f/HEIGHT);
  }
}

// -------- Kernel 2: depthwise conv1d + groupnorm(4) + sigmoid --------
// blockIdx.x = b*8 + which*4 + g ; 256 threads; group = 128 ch x 56 pos = 7168
__global__ void k2_conv_gn_sig(const float* __restrict__ xh, const float* __restrict__ xw,
                               const float* __restrict__ w3, const float* __restrict__ b3,
                               const float* __restrict__ w5, const float* __restrict__ b5,
                               const float* __restrict__ w7, const float* __restrict__ b7,
                               const float* __restrict__ w9, const float* __restrict__ b9,
                               const float* __restrict__ gnh_w, const float* __restrict__ gnh_b,
                               const float* __restrict__ gnw_w, const float* __restrict__ gnw_b,
                               float* __restrict__ ah, float* __restrict__ aw) {
  const int g     = blockIdx.x & 3;
  const int which = (blockIdx.x >> 2) & 1;
  const int b     = blockIdx.x >> 3;
  const float* src = which ? xw : xh;
  float*       dst = which ? aw : ah;
  const float* gw  = which ? gnw_w : gnh_w;
  const float* gb  = which ? gnw_b : gnh_b;
  const float *wk, *bk;
  switch (g) { case 0: wk=w3; bk=b3; break; case 1: wk=w5; bk=b5; break;
               case 2: wk=w7; bk=b7; break; default: wk=w9; bk=b9; break; }
  const int ks = 3 + 2*g, pad = ks >> 1;

  const int NV = (GC*HEIGHT)/256;                  // 28 values per thread
  float vals[NV];
  float s = 0.f, s2 = 0.f;
  for (int i = 0; i < NV; ++i) {
    const int idx = threadIdx.x + i*256;           // 0..7167
    const int cc = idx / HEIGHT, pos = idx % HEIGHT;
    const int c  = g*GC + cc;
    const float* row = src + ((size_t)b*DIM + c)*HEIGHT;
    float acc = bk[cc];
    for (int k = 0; k < ks; ++k) {
      const int p = pos + k - pad;
      if (p >= 0 && p < HEIGHT) acc += row[p] * wk[cc*ks + k];
    }
    vals[i] = acc; s += acc; s2 += acc*acc;
  }
  __shared__ float rs[256], rq[256];
  rs[threadIdx.x] = s; rq[threadIdx.x] = s2;
  __syncthreads();
  for (int off = 128; off > 0; off >>= 1) {
    if ((int)threadIdx.x < off) {
      rs[threadIdx.x] += rs[threadIdx.x+off];
      rq[threadIdx.x] += rq[threadIdx.x+off];
    }
    __syncthreads();
  }
  const float inv_n = 1.f / (float)(GC*HEIGHT);
  const float mu   = rs[0]*inv_n;
  const float rstd = rsqrtf(rq[0]*inv_n - mu*mu + EPS_F);
  for (int i = 0; i < NV; ++i) {
    const int idx = threadIdx.x + i*256;
    const int cc = idx / HEIGHT, pos = idx % HEIGHT;
    const int c  = g*GC + cc;
    const float t = (vals[i]-mu)*rstd*gw[c] + gb[c];
    dst[((size_t)b*DIM + c)*HEIGHT + pos] = 1.f/(1.f + __expf(-t));
  }
}

// -------- Kernel 3: gated 7x7 avg-pool (xg computed on the fly, b128 reads) --------
__global__ void k3_gate_pool(const float* __restrict__ x,
                             const float* __restrict__ ah, const float* __restrict__ aw,
                             float* __restrict__ yraw) {
  __shared__ float pool[LL];
  __shared__ float sah[HEIGHT], saw[WIDTH];
  const int plane = blockIdx.x;                    // b*DIM + c
  const int t = threadIdx.x;
  if (t < LL) pool[t] = 0.f;
  if (t < HEIGHT) sah[t] = ah[(size_t)plane*HEIGHT + t];
  else if (t >= 64 && t < 64 + WIDTH) saw[t-64] = aw[(size_t)plane*WIDTH + (t-64)];
  __syncthreads();
  const f4* src4 = reinterpret_cast<const f4*>(x + (size_t)plane*PLANE);
  for (int i = t; i < PLANE/4; i += 256) {         // 56 % 4 == 0 -> one row per f4
    if (i + 256 < PLANE/4)
      __builtin_prefetch(&src4[i + 256], 0, 1);    // global_prefetch_b8
    const int e0 = i*4, h = e0 / WIDTH, w0 = e0 % WIDTH;
    const float gh = sah[h];
    const int cellBase = (h/WIN)*8;
    f4 v = src4[i];
    #pragma unroll
    for (int j = 0; j < 4; ++j)
      atomicAdd(&pool[cellBase + (w0+j)/WIN], v[j] * gh * saw[w0+j]);  // ds_add_f32
  }
  __syncthreads();
  if (t < LL) yraw[(size_t)plane*LL + t] = pool[t] * (1.f/(WIN*WIN));
}

// -------- Kernel 4: groupnorm(groups=1) over (C,8,8) per batch --------
__global__ void k4_gn_all(const float* __restrict__ yraw,
                          const float* __restrict__ gn_w, const float* __restrict__ gn_b,
                          float* __restrict__ ynorm) {
  __shared__ float rs[1024], rq[1024];
  const int b = blockIdx.x;
  const int N = DIM*LL;                            // 32768
  const float* src = yraw + (size_t)b*N;
  float s = 0.f, s2 = 0.f;
  for (int i = threadIdx.x; i < N; i += 1024) { const float v = src[i]; s += v; s2 += v*v; }
  rs[threadIdx.x] = s; rq[threadIdx.x] = s2;
  __syncthreads();
  for (int off = 512; off > 0; off >>= 1) {
    if ((int)threadIdx.x < off) {
      rs[threadIdx.x] += rs[threadIdx.x+off];
      rq[threadIdx.x] += rq[threadIdx.x+off];
    }
    __syncthreads();
  }
  const float mu   = rs[0]/(float)N;
  const float rstd = rsqrtf(rq[0]/(float)N - mu*mu + EPS_F);
  float* dst = ynorm + (size_t)b*N;
  for (int i = threadIdx.x; i < N; i += 1024) {
    const int c = i >> 6;
    dst[i] = (src[i]-mu)*rstd*gn_w[c] + gn_b[c];
  }
}

// -------- Kernel 5: channel attention via WMMA; emits per-channel gate ca --------
// one block (4 waves) per (b, head). q,k stored [d][l]; v transposed [l][e].
__global__ void k5_attn_wmma(const float* __restrict__ ynorm,
                             const float* __restrict__ qw, const float* __restrict__ kw,
                             const float* __restrict__ vw, float* __restrict__ ca) {
  __shared__ _Float16 qs [64*64];
  __shared__ _Float16 ksh[64*64];
  __shared__ _Float16 vTs[64*64];
  __shared__ _Float16 ash[64*64];
  const int bh = blockIdx.x, b = bh >> 3, head = bh & 7;
  const int tid = threadIdx.x;

  // phase 1: stage q,k,vT as f16
  const float* yb = ynorm + ((size_t)b*DIM + head*HEAD_DIM)*LL;
  for (int i = tid; i < 64*64; i += 128) {
    const int d = i >> 6, l = i & 63;
    const int c = head*HEAD_DIM + d;
    const float yv = yb[(size_t)d*LL + l];
    qs [d*64 + l] = (_Float16)(yv * qw[c]);
    ksh[d*64 + l] = (_Float16)(yv * kw[c]);
    vTs[l*64 + d] = (_Float16)(yv * vw[c]);
  }
  __syncthreads();

  const int wv = tid >> 5, lane = tid & 31;
  const int laneM = lane & 15, hl = lane >> 4;     // ISA A/B/C lane split
  const int dband = wv * 16;

  // phase 2: attn = q @ k^T   (M=d, N=e, K=l)
  v8f acc[4] = {};
  #pragma unroll
  for (int k0 = 0; k0 < 64; k0 += 32) {
    v16h a;
    const _Float16* ar = &qs[(dband + laneM)*64 + k0 + 8*hl];
    #pragma unroll
    for (int j = 0; j < 8; ++j) { a[j] = ar[j]; a[8+j] = ar[16+j]; }
    #pragma unroll
    for (int et = 0; et < 4; ++et) {
      v16h bf;
      const _Float16* br = &ksh[(et*16 + laneM)*64 + k0 + 8*hl];
      #pragma unroll
      for (int j = 0; j < 8; ++j) { bf[j] = br[j]; bf[8+j] = br[16+j]; }
      acc[et] = __builtin_amdgcn_wmma_f32_16x16x32_f16(false, a, false, bf,
                                                       (short)0, acc[et], false, false);
    }
  }
  #pragma unroll
  for (int et = 0; et < 4; ++et) acc[et] *= 0.125f;   // HEAD_DIM^-0.5

  // softmax over e: C-layout row (M) lives in one VGPR across a 16-lane half
  #pragma unroll
  for (int r = 0; r < 8; ++r) {
    float m = fmaxf(fmaxf(acc[0][r], acc[1][r]), fmaxf(acc[2][r], acc[3][r]));
    #pragma unroll
    for (int off = 8; off >= 1; off >>= 1) m = fmaxf(m, __shfl_xor(m, off));
    float e[4], s = 0.f;
    #pragma unroll
    for (int et = 0; et < 4; ++et) { e[et] = __expf(acc[et][r] - m); s += e[et]; }
    #pragma unroll
    for (int off = 8; off >= 1; off >>= 1) s += __shfl_xor(s, off);
    const float inv = 1.f / s;
    const int drow = dband + 8*hl + r;
    #pragma unroll
    for (int et = 0; et < 4; ++et)
      ash[drow*64 + et*16 + laneM] = (_Float16)(e[et] * inv);
  }
  __syncthreads();

  // phase 3: out = attn @ v    (M=d, N=l, K=e)
  v8f oacc[4] = {};
  #pragma unroll
  for (int k0 = 0; k0 < 64; k0 += 32) {
    v16h a;
    const _Float16* ar = &ash[(dband + laneM)*64 + k0 + 8*hl];
    #pragma unroll
    for (int j = 0; j < 8; ++j) { a[j] = ar[j]; a[8+j] = ar[16+j]; }
    #pragma unroll
    for (int lt = 0; lt < 4; ++lt) {
      v16h bf;
      const _Float16* br = &vTs[(lt*16 + laneM)*64 + k0 + 8*hl];
      #pragma unroll
      for (int j = 0; j < 8; ++j) { bf[j] = br[j]; bf[8+j] = br[16+j]; }
      oacc[lt] = __builtin_amdgcn_wmma_f32_16x16x32_f16(false, a, false, bf,
                                                        (short)0, oacc[lt], false, false);
    }
  }
  // per-channel gate: sigmoid(mean over all 64 spatial positions)
  #pragma unroll
  for (int r = 0; r < 8; ++r) {
    float s = oacc[0][r] + oacc[1][r] + oacc[2][r] + oacc[3][r];
    #pragma unroll
    for (int off = 8; off >= 1; off >>= 1) s += __shfl_xor(s, off);
    if (laneM == 0) {
      const int d = dband + 8*hl + r;
      ca[(size_t)b*DIM + head*HEAD_DIM + d] = 1.f/(1.f + __expf(-s * (1.f/64.f)));
    }
  }
}

// -------- Kernel 6: out = ca * x * ah * aw  (recompute xg; NT streaming) --------
__global__ void k6_output(const float* __restrict__ x,
                          const float* __restrict__ ah, const float* __restrict__ aw,
                          const float* __restrict__ ca, float* __restrict__ out) {
  __shared__ float sah[HEIGHT], saw[WIDTH];
  const int plane = blockIdx.x;
  const int t = threadIdx.x;
  if (t < HEIGHT) sah[t] = ah[(size_t)plane*HEIGHT + t];
  else if (t >= 64 && t < 64 + WIDTH) saw[t-64] = aw[(size_t)plane*WIDTH + (t-64)];
  __syncthreads();
  const float cav = ca[plane];
  const f4* src = reinterpret_cast<const f4*>(x   + (size_t)plane*PLANE);
  f4*       dst = reinterpret_cast<f4*>      (out + (size_t)plane*PLANE);
  for (int i = t; i < PLANE/4; i += 256) {           // 56 % 4 == 0 -> same row
    const int e0 = i*4, h = e0 / WIDTH, w0 = e0 % WIDTH;
    const float rh = sah[h] * cav;
    f4 v = __builtin_nontemporal_load(&src[i]);      // last use of x -> NT
    v[0] *= rh * saw[w0+0];
    v[1] *= rh * saw[w0+1];
    v[2] *= rh * saw[w0+2];
    v[3] *= rh * saw[w0+3];
    __builtin_nontemporal_store(v, &dst[i]);         // streaming write, keep L2 clean
  }
}

extern "C" void kernel_launch(void* const* d_in, const int* in_sizes, int n_in,
                              void* d_out, int out_size, void* d_ws, size_t ws_size,
                              hipStream_t stream) {
  (void)in_sizes; (void)n_in; (void)out_size; (void)ws_size;
  const float* x     = (const float*)d_in[0];
  const float* w3    = (const float*)d_in[1];
  const float* b3    = (const float*)d_in[2];
  const float* w5    = (const float*)d_in[3];
  const float* b5    = (const float*)d_in[4];
  const float* w7    = (const float*)d_in[5];
  const float* b7    = (const float*)d_in[6];
  const float* w9    = (const float*)d_in[7];
  const float* b9    = (const float*)d_in[8];
  const float* gnh_w = (const float*)d_in[9];
  const float* gnh_b = (const float*)d_in[10];
  const float* gnw_w = (const float*)d_in[11];
  const float* gnw_b = (const float*)d_in[12];
  const float* gn_w  = (const float*)d_in[13];
  const float* gn_b  = (const float*)d_in[14];
  const float* qw    = (const float*)d_in[15];
  const float* kw    = (const float*)d_in[16];
  const float* vw    = (const float*)d_in[17];
  float* out = (float*)d_out;

  float* ws = (float*)d_ws;
  const size_t SZ_AXIS = (size_t)BATCH*DIM*HEIGHT;   // 917504 floats
  const size_t SZ_Y    = (size_t)BATCH*DIM*LL;       // 1048576 floats
  float* xh    = ws;
  float* xw    = xh   + SZ_AXIS;
  float* ahat  = xw   + SZ_AXIS;
  float* awat  = ahat + SZ_AXIS;
  float* yraw  = awat + SZ_AXIS;
  float* ynorm = yraw + SZ_Y;
  float* cav   = ynorm + SZ_Y;                       // BATCH*DIM floats

  const int NP = BATCH*DIM;                          // 16384 planes
  hipLaunchKernelGGL(k1_axis_means, dim3(NP), dim3(128), 0, stream, x, xh, xw);
  hipLaunchKernelGGL(k2_conv_gn_sig, dim3(BATCH*8), dim3(256), 0, stream,
                     xh, xw, w3,b3, w5,b5, w7,b7, w9,b9,
                     gnh_w,gnh_b, gnw_w,gnw_b, ahat, awat);
  hipLaunchKernelGGL(k3_gate_pool, dim3(NP), dim3(256), 0, stream, x, ahat, awat, yraw);
  hipLaunchKernelGGL(k4_gn_all, dim3(BATCH), dim3(1024), 0, stream, yraw, gn_w, gn_b, ynorm);
  hipLaunchKernelGGL(k5_attn_wmma, dim3(BATCH*HEADS), dim3(128), 0, stream, ynorm, qw, kw, vw, cav);
  hipLaunchKernelGGL(k6_output, dim3(NP), dim3(256), 0, stream, x, ahat, awat, cav, out);
}
